// Memory_69148973466376
// MI455X (gfx1250) — matrix operand order repaired and checked by reading, not compiled, not run on hardware
//
#include <hip/hip_runtime.h>
#include <math.h>

#define B_N 4
#define D_E 128
#define D_O 512
#define HW  1024
#define THW 8192
#define KSPLIT 4
#define KSEG (THW / KSPLIT)   // 2048

typedef _Float16 h8   __attribute__((ext_vector_type(8)));
typedef _Float16 h4   __attribute__((ext_vector_type(4)));
typedef _Float16 v16h __attribute__((ext_vector_type(16)));
typedef float    v8f  __attribute__((ext_vector_type(8)));
typedef float    f4v  __attribute__((ext_vector_type(4)));

__device__ __forceinline__ v16h cat16(h8 lo, h8 hi) {
  return __builtin_shufflevector(lo, hi, 0,1,2,3,4,5,6,7,8,9,10,11,12,13,14,15);
}

// ---------------------------------------------------------------------------
// Transpose + convert:  src f32 [B][D][K]  ->  dst f16 [B][K][D]
// LDS 64x64 tile, coalesced on both sides.
// ---------------------------------------------------------------------------
__global__ __launch_bounds__(256) void k_cvt_T(
    const float* __restrict__ src, _Float16* __restrict__ dst,
    int D, int K)
{
  const int k0 = blockIdx.x * 64;
  const int d0 = blockIdx.y * 64;
  const int b  = blockIdx.z;
  const int tid = threadIdx.x;
  __shared__ _Float16 tile[64 * 72];
  const float* s = src + (size_t)b * D * K;
  _Float16*    o = dst + (size_t)b * K * D;
  const int rr = tid >> 4;          // 0..15
  const int cc = (tid & 15) << 2;   // 0..60
#pragma unroll
  for (int p = 0; p < 4; ++p) {
    const int dr = p * 16 + rr;
    const f4v x = *(const f4v*)&s[(size_t)(d0 + dr) * K + k0 + cc];
    h4 hx; hx.x = (_Float16)x.x; hx.y = (_Float16)x.y;
    hx.z = (_Float16)x.z; hx.w = (_Float16)x.w;
    *(h4*)&tile[dr * 72 + cc] = hx;
  }
  __syncthreads();
#pragma unroll
  for (int p = 0; p < 4; ++p) {
    const int kr = p * 16 + rr;
    h4 hv;
    hv.x = tile[(cc + 0) * 72 + kr];
    hv.y = tile[(cc + 1) * 72 + kr];
    hv.z = tile[(cc + 2) * 72 + kr];
    hv.w = tile[(cc + 3) * 72 + kr];
    *(h4*)&o[(size_t)(k0 + kr) * D + d0 + cc] = hv;
  }
}

// Elementwise convert: m_val f32 -> f16 (same [d][k] layout).
__global__ __launch_bounds__(256) void k_cvt(
    const float* __restrict__ src, _Float16* __restrict__ dst)
{
  const size_t i = ((size_t)blockIdx.x * 256 + threadIdx.x) * 4;
  const f4v x = *(const f4v*)&src[i];
  h4 hx; hx.x = (_Float16)x.x; hx.y = (_Float16)x.y;
  hx.z = (_Float16)x.z; hx.w = (_Float16)x.w;
  *(h4*)&dst[i] = hx;
}

// ---------------------------------------------------------------------------
// GEMM-1 (f16 WMMA 16x16x32) + distance bias + fused softmax partial stats.
// Tile 128k x 64q per 256-thread WG; wave owns 16k x 64q (4 acc tiles).
// A = m_keyT [k][d] (contiguous K runs), B = q_keyT [q][d].
// Epilogue writes raw f32 scores and per-(ktile,q) (max, sum-exp) partials.
// ---------------------------------------------------------------------------
__global__ __launch_bounds__(256) void k_scores(
    const _Float16* __restrict__ mkT,   // [B][THW][D_E]
    const _Float16* __restrict__ qkT,   // [B][HW][D_E]
    float* __restrict__ out_p,          // raw scores (B,THW,HW)
    float* __restrict__ spart)          // [B*HW][64 ktiles][2]
{
  const int qt = blockIdx.x;   // 16
  const int kt = blockIdx.y;   // 64
  const int b  = blockIdx.z;
  const int qbase = qt * 64, kbase = kt * 128;
  const int tid = threadIdx.x, wave = tid >> 5, lane = tid & 31;
  const int l15 = lane & 15, hi = lane >> 4;

  __shared__ float red_m[8][64];
  __shared__ float red_s[8][64];

  const _Float16* aT = mkT + (size_t)b * THW * D_E;
  const _Float16* bT = qkT + (size_t)b * HW * D_E;

  v8f acc[4] = {};
  const int arow = kbase + wave * 16 + l15;
#pragma unroll
  for (int dc = 0; dc < D_E; dc += 32) {
    // A frag 16x32: lane<16 K runs [dc..dc+8)+[dc+16..dc+24); lane>=16 +8
    const _Float16* ap = aT + (size_t)arow * D_E + dc + hi * 8;
    const v16h a = cat16(*(const h8*)ap, *(const h8*)(ap + 16));
#pragma unroll
    for (int j = 0; j < 4; ++j) {
      // B frag 32x16: lane<16 K [dc..dc+8)+[dc+16..dc+24); lane>=16 +8
      const _Float16* bp =
          bT + (size_t)(qbase + j * 16 + l15) * D_E + dc + hi * 8;
      const v16h bb = cat16(*(const h8*)bp, *(const h8*)(bp + 16));
      acc[j] = __builtin_amdgcn_wmma_f32_16x16x32_f16(
          false, a, false, bb, (short)0, acc[j], false, false);
    }
  }

  const float scale = 0.088388347648318447f;  // 1/sqrt(128)
  float jm[4], js[4];
#pragma unroll
  for (int j = 0; j < 4; ++j) {
    const int q = qbase + j * 16 + l15;
    const float qy = (float)(q >> 5), qx = (float)(q & 31);
    float xv[8];
#pragma unroll
    for (int v = 0; v < 8; ++v) {
      const int k   = kbase + wave * 16 + hi * 8 + v;
      const int khw = k & (HW - 1);
      const float dy = (float)(khw >> 5) - qy;
      const float dx = (float)(khw & 31) - qx;
      const float dm = sqrtf(dy * dy + dx * dx) + 1.0f;
      xv[v] = acc[j][v] * (scale / dm);
      out_p[((size_t)b * THW + k) * HW + q] = xv[v];
    }
    float m = xv[0];
#pragma unroll
    for (int v = 1; v < 8; ++v) m = fmaxf(m, xv[v]);
    float s = 0.0f;
#pragma unroll
    for (int v = 0; v < 8; ++v) s += __expf(xv[v] - m);
    // merge lane L with lane L^16 (same q, other 8 k rows)
    const float mo = __shfl_xor(m, 16, 32);
    const float so = __shfl_xor(s, 16, 32);
    const float mn = fmaxf(m, mo);
    s = s * __expf(m - mn) + so * __expf(mo - mn);
    jm[j] = mn; js[j] = s;
  }
  if (hi == 0) {
#pragma unroll
    for (int j = 0; j < 4; ++j) {
      red_m[wave][j * 16 + l15] = jm[j];
      red_s[wave][j * 16 + l15] = js[j];
    }
  }
  __syncthreads();
  if (tid < 64) {   // one thread per q column: merge 8 waves
    float m = red_m[0][tid], s = red_s[0][tid];
#pragma unroll
    for (int w = 1; w < 8; ++w) {
      const float pm = red_m[w][tid], ps = red_s[w][tid];
      const float mn = fmaxf(m, pm);
      s = s * __expf(m - mn) + ps * __expf(pm - mn);
      m = mn;
    }
    const size_t qg = (size_t)b * HW + qbase + tid;
    spart[(qg * 64 + kt) * 2]     = m;
    spart[(qg * 64 + kt) * 2 + 1] = s;
  }
}

// Merge 64 k-tile partials -> SoA (col_max, 1/col_sum) per (b,q).
__global__ __launch_bounds__(256) void k_smax_merge(
    const float* __restrict__ spart,
    float* __restrict__ smax, float* __restrict__ sinv)
{
  const int i = blockIdx.x * 256 + threadIdx.x;   // b*HW + q (exact 4096)
  float m = -3.0e38f, s = 0.0f;
  for (int kt = 0; kt < 64; ++kt) {
    const float pm = spart[((size_t)i * 64 + kt) * 2];
    const float ps = spart[((size_t)i * 64 + kt) * 2 + 1];
    const float mn = fmaxf(m, pm);
    s = s * __expf(m - mn) + ps * __expf(pm - mn);
    m = mn;
  }
  smax[i] = m;
  sinv[i] = 1.0f / s;
}

// ---------------------------------------------------------------------------
// GEMM-2 (f16 WMMA) fused with softmax normalization + final-p write-back.
// WG tile: full 512 d x 64 q over one k segment (THW/4); 256 WGs total.
// Each raw-p chunk owned by exactly one WG -> in-place normalize is race-free.
// Double-buffered LDS (one barrier per iteration): next chunk's global loads
// overlap the current chunk's WMMAs.
// ---------------------------------------------------------------------------
#define AST 40   // halves; 80B rows: 16B-aligned b128 frags, conflict-free
#define PST 40

__global__ __launch_bounds__(256) void k_attend(
    const _Float16* __restrict__ mvh,   // [B][512][8192] f16
    const float* __restrict__ smax, const float* __restrict__ sinv,
    float* __restrict__ p_io, float* __restrict__ pmem)
{
  const int qs = blockIdx.x;   // 16 strips of 64 q
  const int ks = blockIdx.y;   // 4 k segments
  const int b  = blockIdx.z;
  const int qbase = qs * 64;
  const int kbeg  = ks * KSEG;
  const int tid = threadIdx.x, wave = tid >> 5, lane = tid & 31;
  const int l15 = lane & 15, hi = lane >> 4;

  __shared__ _Float16 a_s[2][D_O * AST];   // 2 x 40 KB
  __shared__ _Float16 p_sT[2][64 * PST];   // 2 x  5 KB

  const _Float16* mo = mvh + (size_t)b * D_O * THW;
  float* pp = p_io + (size_t)b * THW * HW;

  const int tr = tid >> 4;          // k row 0..15
  const int tc = (tid & 15) << 2;   // q quad 0..60
  const f4v cm = *(const f4v*)&smax[(size_t)b * HW + qbase + tc];
  const f4v ci = *(const f4v*)&sinv[(size_t)b * HW + qbase + tc];

  // Stage chunk [kc..kc+32): normalize p (NT write-back + transposed f16 to
  // LDS) and copy the m_val slab.
  auto stage = [&](int kc, int buf) {
#pragma unroll
    for (int p2 = 0; p2 < 2; ++p2) {
      const int r = p2 * 16 + tr;
      float* g = &pp[(size_t)(kc + r) * HW + qbase + tc];
      const f4v x = *(const f4v*)g;
      f4v pn;
      pn.x = __expf(x.x - cm.x) * ci.x;
      pn.y = __expf(x.y - cm.y) * ci.y;
      pn.z = __expf(x.z - cm.z) * ci.z;
      pn.w = __expf(x.w - cm.w) * ci.w;
      __builtin_nontemporal_store(pn, (f4v*)g);
      p_sT[buf][(tc + 0) * PST + r] = (_Float16)pn.x;
      p_sT[buf][(tc + 1) * PST + r] = (_Float16)pn.y;
      p_sT[buf][(tc + 2) * PST + r] = (_Float16)pn.z;
      p_sT[buf][(tc + 3) * PST + r] = (_Float16)pn.w;
    }
#pragma unroll
    for (int i = 0; i < 8; ++i) {
      const int d  = i * 64 + (tid >> 2);
      const int ko = (tid & 3) << 3;
      *(h8*)&a_s[buf][d * AST + ko] =
          *(const h8*)&mo[(size_t)d * THW + kc + ko];
    }
  };

  v8f acc[4][4] = {};

  stage(kbeg, 0);
  int cur = 0;
  for (int kc = kbeg; kc < kbeg + KSEG; kc += 32) {
    __syncthreads();   // buf[cur] staged; buf[cur^1] fully consumed
    if (kc + 32 < kbeg + KSEG) stage(kc + 32, cur ^ 1);   // overlaps WMMAs

    v16h afr[4];
#pragma unroll
    for (int td = 0; td < 4; ++td) {
      const _Float16* ap =
          &a_s[cur][(wave * 64 + td * 16 + l15) * AST + hi * 8];
      afr[td] = cat16(*(const h8*)ap, *(const h8*)(ap + 16));
    }
    v16h bfr[4];
#pragma unroll
    for (int tn = 0; tn < 4; ++tn) {
      const _Float16* bp = &p_sT[cur][(tn * 16 + l15) * PST + hi * 8];
      bfr[tn] = cat16(*(const h8*)bp, *(const h8*)(bp + 16));
    }
#pragma unroll
    for (int td = 0; td < 4; ++td)
#pragma unroll
      for (int tn = 0; tn < 4; ++tn)
        acc[td][tn] = __builtin_amdgcn_wmma_f32_16x16x32_f16(
            false, afr[td], false, bfr[tn], (short)0, acc[td][tn],
            false, false);
    cur ^= 1;
  }

  float* po = pmem + ((size_t)(b * KSPLIT + ks)) * D_O * HW;
#pragma unroll
  for (int td = 0; td < 4; ++td)
#pragma unroll
    for (int tn = 0; tn < 4; ++tn) {
      const int q = qbase + tn * 16 + l15;
#pragma unroll
      for (int v = 0; v < 8; ++v) {
        const int d = wave * 64 + td * 16 + hi * 8 + v;
        po[(size_t)d * HW + q] = acc[td][tn][v];
      }
    }
}

// Sum KSPLIT partials -> out_mem rows d<512 (NT stores).
__global__ __launch_bounds__(256) void k_reduce_mem(
    const float* __restrict__ pmem, float* __restrict__ out_mem)
{
  const size_t i = ((size_t)blockIdx.x * 256 + threadIdx.x) * 4;
  const size_t b = i >> 19;                  // D_O*HW = 2^19
  const size_t r = i & (((size_t)1 << 19) - 1);
  f4v s = *(const f4v*)&pmem[((b * KSPLIT) << 19) + r];
#pragma unroll
  for (int ks = 1; ks < KSPLIT; ++ks) {
    const f4v x = *(const f4v*)&pmem[((b * KSPLIT + ks) << 19) + r];
    s.x += x.x; s.y += x.y; s.z += x.z; s.w += x.w;
  }
  __builtin_nontemporal_store(s, (f4v*)&out_mem[(b << 20) + r]);
}

// Concat copy: mem_val[:, 512:1024] = q_val  (float4, NT).
__global__ __launch_bounds__(256) void k_copy_qval(
    const float* __restrict__ qval, float* __restrict__ out_mem)
{
  const size_t i = ((size_t)blockIdx.x * 256 + threadIdx.x) * 4;
  const size_t b = i >> 19;
  const size_t r = i & (((size_t)1 << 19) - 1);
  const f4v x = *(const f4v*)&qval[i];
  __builtin_nontemporal_store(
      x, (f4v*)&out_mem[(b << 20) + ((size_t)1 << 19) + r]);
}

// ---------------------------------------------------------------------------
extern "C" void kernel_launch(void* const* d_in, const int* in_sizes, int n_in,
                              void* d_out, int out_size, void* d_ws,
                              size_t ws_size, hipStream_t stream) {
  const float* mkey = (const float*)d_in[0];   // (B, D_E, THW)
  const float* mval = (const float*)d_in[1];   // (B, D_O, THW)
  const float* qkey = (const float*)d_in[2];   // (B, D_E, HW)
  const float* qval = (const float*)d_in[3];   // (B, D_O, HW)

  float* out_mem = (float*)d_out;                          // (B, 2*D_O, HW)
  float* out_p   = out_mem + (size_t)B_N * 2 * D_O * HW;   // (B, THW, HW)

  char* w = (char*)d_ws;
  _Float16* mkT = (_Float16*)w; w += (size_t)B_N * THW * D_E * 2;      // 8.4 MB
  _Float16* qkT = (_Float16*)w; w += (size_t)B_N * HW * D_E * 2;       // 1 MB
  _Float16* mvh = (_Float16*)w; w += (size_t)B_N * D_O * THW * 2;      // 33.5 MB
  float* pmem   = (float*)w;    w += (size_t)B_N * KSPLIT * D_O * HW * 4; // 33.5 MB
  float* spart  = (float*)w;    w += (size_t)B_N * HW * 64 * 2 * 4;    // 2 MB
  float* smax   = (float*)w;    w += (size_t)B_N * HW * 4;
  float* sinv   = (float*)w;    w += (size_t)B_N * HW * 4;

  k_cvt_T<<<dim3(THW / 64, D_E / 64, B_N), 256, 0, stream>>>(mkey, mkT, D_E, THW);
  k_cvt_T<<<dim3(HW / 64, D_E / 64, B_N), 256, 0, stream>>>(qkey, qkT, D_E, HW);
  k_cvt<<<dim3((B_N * D_O * THW) / 1024), 256, 0, stream>>>(mval, mvh);

  k_scores<<<dim3(16, 64, B_N), 256, 0, stream>>>(mkT, qkT, out_p, spart);
  k_smax_merge<<<dim3(16), 256, 0, stream>>>(spart, smax, sinv);
  k_attend<<<dim3(16, KSPLIT, B_N), 256, 0, stream>>>(mvh, smax, sinv, out_p, pmem);
  k_reduce_mem<<<dim3(2048), 256, 0, stream>>>(pmem, out_mem);
  k_copy_qval<<<dim3(2048), 256, 0, stream>>>(qval, out_mem);
}
